// CombinedAttentionProcessor_89575837925491
// MI455X (gfx1250) — matrix operand order repaired and checked by reading, not compiled, not run on hardware
//
#include <hip/hip_runtime.h>

// ---------------------------------------------------------------------------
// CDNA5 (gfx1250) fused self-attention block:
//   QKV proj (WMMA f16 GEMM) -> flash attention (WMMA + online softmax,
//   denominator fused into the WMMA accumulator via a ones-row in V^T)
//   -> output proj + bias + residual (WMMA f16 GEMM, f32 epilogue)
// ---------------------------------------------------------------------------

typedef __attribute__((ext_vector_type(16))) _Float16 v16h;
typedef __attribute__((ext_vector_type(8)))  _Float16 v8h;
typedef __attribute__((ext_vector_type(8)))  float    v8f;

union V16U { v16h v; v8h h[2]; };

static constexpr int B_   = 2;
static constexpr int S_   = 4096;
static constexpr int C_   = 640;
static constexpr int H_   = 8;
static constexpr int D_   = 80;   // head dim
static constexpr int DP_  = 96;   // padded head dim (3 x 32)
static constexpr int VTR_ = 96;   // V^T rows: 80 data + 1 ones + 15 zero
static constexpr int BH_  = B_ * H_;        // 16
static constexpr int M_   = B_ * S_;        // 8192 tokens

// ---------------------------------------------------------------------------
// WMMA fragment loaders (ISA 7.12.2 layouts), ldk = row stride in halves.
// A (16x32 f16): lane L holds row M=L%16; halves 0..7 = K=(L>=16?8:0)+0..7,
//                halves 8..15 = K=16+(L>=16?8:0)+0..7.
// B (32x16 f16): lane L holds col N=L%16; halves 0..7 = K=(L>=16?8:0)+0..7,
//                halves 8..15 = K=16+(L>=16?8:0)+0..7.
// C (16x16 f32): VGPR r, lane L -> M = r + (L>=16?8:0), N = L%16.
// ---------------------------------------------------------------------------
__device__ __forceinline__ v16h load_a_frag(const _Float16* __restrict__ base, int ldk) {
  const int lane = threadIdx.x & 31;
  const int row  = lane & 15;
  const int sh   = (lane >> 4) << 3;         // 0 or 8
  const _Float16* p = base + row * ldk + sh;
  V16U r;
  r.h[0] = *(const v8h*)(p);
  r.h[1] = *(const v8h*)(p + 16);
  return r.v;
}

__device__ __forceinline__ v16h load_b_frag(const _Float16* __restrict__ base, int ldk) {
  const int lane = threadIdx.x & 31;
  const int col  = lane & 15;
  const int kb   = (lane >> 4) << 3;         // 0 or 8
  const _Float16* p = base + col * ldk + kb;
  V16U r;
  r.h[0] = *(const v8h*)(p);
  r.h[1] = *(const v8h*)(p + 16);
  return r.v;
}

__device__ __forceinline__ v8f wmma_f16(v16h a, v16h b, v8f c) {
  return __builtin_amdgcn_wmma_f32_16x16x32_f16(false, a, false, b, (short)0, c, false, false);
}

// ---------------------------------------------------------------------------
// Max-reduce + broadcast across the 16 lanes of a half-wave.
// v_permlane16_b32 xor-butterfly (pure VALU, no DS waits), stages 1,2,4,8.
// ---------------------------------------------------------------------------
__device__ __forceinline__ float halfwave_max16(float v) {
#if __has_builtin(__builtin_amdgcn_permlane16)
  union FU { float f; unsigned u; } a, b;
  a.f = v;
  b.u = __builtin_amdgcn_permlane16(a.u, a.u, 0x67452301u, 0xEFCDAB89u, false, false);
  a.f = fmaxf(a.f, b.f);
  b.u = __builtin_amdgcn_permlane16(a.u, a.u, 0x54761032u, 0xDCFE98BAu, false, false);
  a.f = fmaxf(a.f, b.f);
  b.u = __builtin_amdgcn_permlane16(a.u, a.u, 0x32107654u, 0xBA98FEDCu, false, false);
  a.f = fmaxf(a.f, b.f);
  b.u = __builtin_amdgcn_permlane16(a.u, a.u, 0xFEDCBA98u, 0x76543210u, false, false);
  a.f = fmaxf(a.f, b.f);
  return a.f;
#else
  v = fmaxf(v, __shfl_xor(v, 1, 32));
  v = fmaxf(v, __shfl_xor(v, 2, 32));
  v = fmaxf(v, __shfl_xor(v, 4, 32));
  v = fmaxf(v, __shfl_xor(v, 8, 32));
  return v;
#endif
}

// ---------------------------------------------------------------------------
// Init kernel: zero padded dims (80..95) of Q/K rows, and build the V^T pad
// rows: row 80 = 1.0 (softmax-denominator column), rows 81..95 = 0.
// ---------------------------------------------------------------------------
__global__ void init_pads_kernel(_Float16* __restrict__ qbuf,
                                 _Float16* __restrict__ kbuf,
                                 _Float16* __restrict__ vt) {
  const int i = blockIdx.x * blockDim.x + threadIdx.x;
  const int qkRows = BH_ * S_;                 // 65536 rows per buffer
  const int vtWork = BH_ * 16 * (S_ / 16);     // 16 pad rows, 16 halves/thread

  if (i < 2 * qkRows) {
    _Float16* base = (i < qkRows) ? qbuf : kbuf;
    const int row  = (i < qkRows) ? i : (i - qkRows);
    uint32_t* p = (uint32_t*)(base + (size_t)row * DP_ + D_);
#pragma unroll
    for (int j = 0; j < 8; ++j) p[j] = 0u;
  } else if (i < 2 * qkRows + vtWork) {
    const int j   = i - 2 * qkRows;
    const int bh  = j / (16 * (S_ / 16));
    const int rem = j - bh * (16 * (S_ / 16));
    const int dr  = rem / (S_ / 16);           // 0..15 -> dim 80+dr
    const int seg = rem - dr * (S_ / 16);      // 0..255
    const uint32_t val = (dr == 0) ? 0x3C003C00u : 0u;   // packed f16 {1,1}/{0,0}
    uint32_t* p = (uint32_t*)(vt + ((size_t)bh * VTR_ + D_ + dr) * S_ + seg * 16);
#pragma unroll
    for (int k = 0; k < 8; ++k) p[k] = val;
  }
}

// ---------------------------------------------------------------------------
// QKV projection:  dst = X @ W^T  (X: [8192,640] f32, W: [640,640] f32)
// MODE 0: Q -> qbuf[bh][s][96] f16, pre-scaled by 1/sqrt(D)
// MODE 1: K -> kbuf[bh][s][96] f16
// MODE 2: V -> vt  [bh][96][4096] f16 (transposed for AV B-fragments)
// Block: 128 threads (4 waves). Block tile 64x64, wave tile 16x64.
// ---------------------------------------------------------------------------
template <int MODE>
__global__ void __launch_bounds__(128)
qkv_gemm_kernel(const float* __restrict__ X,
                const float* __restrict__ W,
                _Float16* __restrict__ dst) {
  __shared__ _Float16 As[64 * 32];
  __shared__ _Float16 Ws[64 * 32];

  const int t    = threadIdx.x;
  const int wv   = t >> 5;
  const int lane = t & 31;
  const int m0   = blockIdx.y * 64;
  const int n0   = blockIdx.x * 64;

  v8f acc[4] = {};

  const int lrow = t >> 1;             // 0..63
  const int lks  = (t & 1) * 16;       // 0 or 16

  for (int k0 = 0; k0 < C_; k0 += 32) {
    {
      const float4* src = (const float4*)(X + (size_t)(m0 + lrow) * C_ + k0 + lks);
      _Float16* d = As + lrow * 32 + lks;
#pragma unroll
      for (int i = 0; i < 4; ++i) {
        float4 f = src[i];
        d[i * 4 + 0] = (_Float16)f.x;
        d[i * 4 + 1] = (_Float16)f.y;
        d[i * 4 + 2] = (_Float16)f.z;
        d[i * 4 + 3] = (_Float16)f.w;
      }
    }
    {
      const float4* src = (const float4*)(W + (size_t)(n0 + lrow) * C_ + k0 + lks);
      _Float16* d = Ws + lrow * 32 + lks;
#pragma unroll
      for (int i = 0; i < 4; ++i) {
        float4 f = src[i];
        d[i * 4 + 0] = (_Float16)f.x;
        d[i * 4 + 1] = (_Float16)f.y;
        d[i * 4 + 2] = (_Float16)f.z;
        d[i * 4 + 3] = (_Float16)f.w;
      }
    }
    __syncthreads();

    v16h a = load_a_frag(As + wv * 16 * 32, 32);
#pragma unroll
    for (int nt = 0; nt < 4; ++nt) {
      v16h b = load_b_frag(Ws + nt * 16 * 32, 32);
      acc[nt] = wmma_f16(a, b, acc[nt]);
    }
    __syncthreads();
  }

  const int nl = lane & 15;
  const int mh = (lane >> 4) << 3;
  const float qscale = 0.11180339887498949f;  // 1/sqrt(80)
#pragma unroll
  for (int nt = 0; nt < 4; ++nt) {
#pragma unroll
    for (int r = 0; r < 8; ++r) {
      const int row = m0 + wv * 16 + r + mh;  // token 0..8191
      const int col = n0 + nt * 16 + nl;      // 0..639
      const int b   = row >> 12;              // /4096
      const int s   = row & (S_ - 1);
      const int h   = col / D_;
      const int d   = col - h * D_;
      const int bh  = b * H_ + h;
      const float v = acc[nt][r];
      if (MODE == 0) {
        dst[((size_t)bh * S_ + s) * DP_ + d] = (_Float16)(v * qscale);
      } else if (MODE == 1) {
        dst[((size_t)bh * S_ + s) * DP_ + d] = (_Float16)v;
      } else {
        dst[((size_t)bh * VTR_ + d) * S_ + s] = (_Float16)v;
      }
    }
  }
}

// ---------------------------------------------------------------------------
// Flash attention core. One wave per (bh, 16-query tile).
// __launch_bounds__(32,1) + waves_per_eu(1): single-wave block gets the full
// VGPR budget so Q fragments + 6 accumulators stay resident (no scratch
// spills/reloads in the K-loop; latency hidden by 4096 independent blocks).
// ---------------------------------------------------------------------------
__global__ void __launch_bounds__(32, 1) __attribute__((amdgpu_waves_per_eu(1)))
flash_attn_kernel(const _Float16* __restrict__ qbuf,
                  const _Float16* __restrict__ kbuf,
                  const _Float16* __restrict__ vt,
                  _Float16* __restrict__ attnout) {
  const int qt   = blockIdx.x;   // 0..255
  const int bh   = blockIdx.y;   // 0..15
  const int lane = threadIdx.x & 31;

  const _Float16* qb = qbuf + (size_t)bh * S_ * DP_ + (size_t)qt * 16 * DP_;
  const _Float16* kb = kbuf + (size_t)bh * S_ * DP_;
  const _Float16* vb = vt   + (size_t)bh * VTR_ * S_;

  __shared__ _Float16 pbuf[16 * 32];   // P tile staged C-layout -> A-layout

  v16h qf[3];
#pragma unroll
  for (int c = 0; c < 3; ++c) qf[c] = load_a_frag(qb + c * 32, DP_);

  float mrow[8];
#pragma unroll
  for (int r = 0; r < 8; ++r) mrow[r] = -3.0e38f;
  v8f acc[6] = {};   // 5 output tiles (16x80) + 1 denominator tile (col 0)

  for (int kt = 0; kt < S_ / 32; ++kt) {
    const _Float16* kbase = kb + (size_t)kt * 32 * DP_;

    // prefetch next K tile and next V stripe into cache (speculative)
    __builtin_prefetch(kbase + 32 * DP_, 0, 1);
    __builtin_prefetch(vb + (kt + 1) * 32, 0, 1);

    v8f s0 = {};   // scores vs keys kt*32 + 0..15
    v8f s1 = {};   // scores vs keys kt*32 + 16..31
#pragma unroll
    for (int c = 0; c < 3; ++c) {
      v16h b0 = load_b_frag(kbase + c * 32, DP_);
      v16h b1 = load_b_frag(kbase + 16 * DP_ + c * 32, DP_);
      s0 = wmma_f16(qf[c], b0, s0);
      s1 = wmma_f16(qf[c], b1, s1);
    }

    // online softmax: per-row running max (VALU permlane reduction);
    // the denominator is carried by the WMMA accumulator acc[5].
#pragma unroll
    for (int r = 0; r < 8; ++r) {
      const float v     = halfwave_max16(fmaxf(s0[r], s1[r]));
      const float mn    = fmaxf(mrow[r], v);
      const float alpha = __expf(mrow[r] - mn);
      const float p0    = __expf(s0[r] - mn);
      const float p1    = __expf(s1[r] - mn);
      mrow[r] = mn;
#pragma unroll
      for (int nt = 0; nt < 6; ++nt) acc[nt][r] *= alpha;
      const int prow = r + ((lane >> 4) << 3);
      const int pcol = lane & 15;
      pbuf[prow * 32 + pcol]      = (_Float16)p0;
      pbuf[prow * 32 + 16 + pcol] = (_Float16)p1;
    }
    __syncthreads();

    v16h pf = load_a_frag(pbuf, 32);
#pragma unroll
    for (int nt = 0; nt < 6; ++nt) {
      v16h vf = load_b_frag(vb + (size_t)(nt * 16) * S_ + kt * 32, S_);
      acc[nt] = wmma_f16(pf, vf, acc[nt]);
    }
    __syncthreads();
  }

  // epilogue: l for row r(+8) sits in acc[5] column 0 = lanes 0 / 16.
  float linv[8];
#pragma unroll
  for (int r = 0; r < 8; ++r) {
    const float l = __shfl(acc[5][r], lane & 16, 32);
    linv[r] = 1.0f / l;
  }

  const int b  = bh >> 3;
  const int h  = bh & 7;
  const int nl = lane & 15;
  const int mh = (lane >> 4) << 3;
#pragma unroll
  for (int nt = 0; nt < 5; ++nt) {
#pragma unroll
    for (int r = 0; r < 8; ++r) {
      const int s = qt * 16 + r + mh;
      const int d = nt * 16 + nl;
      const float o = acc[nt][r] * linv[r];
      attnout[((size_t)(b * S_ + s)) * C_ + h * D_ + d] = (_Float16)o;
    }
  }
}

// ---------------------------------------------------------------------------
// Output projection + bias + residual: out = A @ Wo^T + bo + X   (f32 out)
// ---------------------------------------------------------------------------
__global__ void __launch_bounds__(128)
out_proj_kernel(const _Float16* __restrict__ A,
                const float* __restrict__ Wo,
                const float* __restrict__ bo,
                const float* __restrict__ X,
                float* __restrict__ out) {
  __shared__ _Float16 As[64 * 32];
  __shared__ _Float16 Ws[64 * 32];

  const int t    = threadIdx.x;
  const int wv   = t >> 5;
  const int lane = t & 31;
  const int m0   = blockIdx.y * 64;
  const int n0   = blockIdx.x * 64;

  v8f acc[4] = {};

  const int lrow = t >> 1;
  const int lks  = (t & 1) * 16;

  for (int k0 = 0; k0 < C_; k0 += 32) {
    {
      const v8h* src = (const v8h*)(A + (size_t)(m0 + lrow) * C_ + k0 + lks);
      *(v8h*)(As + lrow * 32 + lks)     = src[0];
      *(v8h*)(As + lrow * 32 + lks + 8) = src[1];
    }
    {
      const float4* src = (const float4*)(Wo + (size_t)(n0 + lrow) * C_ + k0 + lks);
      _Float16* d = Ws + lrow * 32 + lks;
#pragma unroll
      for (int i = 0; i < 4; ++i) {
        float4 f = src[i];
        d[i * 4 + 0] = (_Float16)f.x;
        d[i * 4 + 1] = (_Float16)f.y;
        d[i * 4 + 2] = (_Float16)f.z;
        d[i * 4 + 3] = (_Float16)f.w;
      }
    }
    __syncthreads();

    v16h a = load_a_frag(As + wv * 16 * 32, 32);
#pragma unroll
    for (int nt = 0; nt < 4; ++nt) {
      v16h b = load_b_frag(Ws + nt * 16 * 32, 32);
      acc[nt] = wmma_f16(a, b, acc[nt]);
    }
    __syncthreads();
  }

  const int nl = lane & 15;
  const int mh = (lane >> 4) << 3;
#pragma unroll
  for (int nt = 0; nt < 4; ++nt) {
#pragma unroll
    for (int r = 0; r < 8; ++r) {
      const int row = m0 + wv * 16 + r + mh;
      const int col = n0 + nt * 16 + nl;
      const size_t idx = (size_t)row * C_ + col;
      out[idx] = acc[nt][r] + bo[col] + X[idx];
    }
  }
}

// ---------------------------------------------------------------------------
// Host launcher
// ---------------------------------------------------------------------------
extern "C" void kernel_launch(void* const* d_in, const int* in_sizes, int n_in,
                              void* d_out, int out_size, void* d_ws, size_t ws_size,
                              hipStream_t stream) {
  (void)in_sizes; (void)n_in; (void)out_size; (void)ws_size;

  const float* hs = (const float*)d_in[0];   // [B,S,C]
  const float* Wq = (const float*)d_in[1];
  const float* Wk = (const float*)d_in[2];
  const float* Wv = (const float*)d_in[3];
  const float* Wo = (const float*)d_in[4];
  const float* bo = (const float*)d_in[5];
  float* out = (float*)d_out;

  // workspace carve-up (f16)
  _Float16* qbuf    = (_Float16*)d_ws;                   // [16][4096][96]
  _Float16* kbuf    = qbuf + (size_t)BH_ * S_ * DP_;     // [16][4096][96]
  _Float16* vtbuf   = kbuf + (size_t)BH_ * S_ * DP_;     // [16][96][4096]
  _Float16* attnout = vtbuf + (size_t)BH_ * VTR_ * S_;   // [8192][640]

  // 1) init pads: zero Q/K pad dims, build ones/zero pad rows of V^T
  {
    const int total = 2 * BH_ * S_ + BH_ * 16 * (S_ / 16);
    init_pads_kernel<<<(total + 255) / 256, 256, 0, stream>>>(qbuf, kbuf, vtbuf);
  }

  // 2) Q/K/V projections (WMMA GEMMs)
  dim3 ggrid(C_ / 64, M_ / 64);   // (10, 128)
  qkv_gemm_kernel<0><<<ggrid, 128, 0, stream>>>(hs, Wq, qbuf);
  qkv_gemm_kernel<1><<<ggrid, 128, 0, stream>>>(hs, Wk, kbuf);
  qkv_gemm_kernel<2><<<ggrid, 128, 0, stream>>>(hs, Wv, vtbuf);

  // 3) flash attention (one wave per 16-query tile)
  flash_attn_kernel<<<dim3(S_ / 16, BH_), 32, 0, stream>>>(qbuf, kbuf, vtbuf, attnout);

  // 4) output projection + bias + residual
  out_proj_kernel<<<ggrid, 128, 0, stream>>>(attnout, Wo, bo, hs, out);
}
